// MeshRCNNGraphConvHead_88416196755964
// MI455X (gfx1250) — matrix operand order repaired and checked by reading, not compiled, not run on hardware
//
#include <hip/hip_runtime.h>
#include <hip/hip_bf16.h>

// Problem constants (from reference)
#define BB   64
#define CC   256
#define HH   14
#define WW   14
#define VV   2562
#define VT_  163968      // BB * VV, divisible by 128
#define EE   491520
#define HIDN 128

typedef __attribute__((ext_vector_type(16))) __bf16        v16bf;
typedef __attribute__((ext_vector_type(8)))  float         v8f;
typedef __attribute__((ext_vector_type(8)))  unsigned int  v8u;

__device__ __forceinline__ unsigned short f2bf(float f) {
    unsigned int u = __builtin_bit_cast(unsigned int, f);
    u += 0x7fffu + ((u >> 16) & 1u);           // round-to-nearest-even
    return (unsigned short)(u >> 16);
}
__device__ __forceinline__ float bf2f(unsigned short h) {
    unsigned int u = ((unsigned int)h) << 16;
    return __builtin_bit_cast(float, u);
}

// ---------------------------------------------------------------------------
// WMMA GEMM: C[M x 128] = A[M x K] * W^T + bias, A bf16 (row stride As),
// W stored N-major bf16 [128 x Kp] (== PyTorch (out,in) layout, K zero-padded).
// Block = 256 threads = 8 waves; each wave computes a 16 x 128 strip.
// MODE 0: store fp32 to out[m*128+n].  MODE 1: relu -> bf16 at out[m*ostride+ocol+n].
// ---------------------------------------------------------------------------
template<int MODE>
__global__ void __launch_bounds__(256)
gemm_bf16_wmma(const unsigned short* __restrict__ A, int As,
               const unsigned short* __restrict__ Bw, int Kp,
               const float* __restrict__ bias,
               void* out, int ostride, int ocol)
{
    const int wave  = threadIdx.x >> 5;
    const int lane  = threadIdx.x & 31;
    const int lhalf = lane >> 4;            // 0: lanes 0-15, 1: lanes 16-31
    const int l16   = lane & 15;
    const int mbase = blockIdx.x * 128 + wave * 16;

    // A-fragment K offsets: lane-half selects K 0-15 vs 16-31 (pairs of bf16 per dword)
    int akofs[8];
#pragma unroll
    for (int d = 0; d < 8; ++d) {
        akofs[d] = lhalf * 16 + 2 * d;
    }
    // B-fragment K offsets: lane-half selects K 0-15 vs 16-31
    int bkofs[8];
#pragma unroll
    for (int d = 0; d < 8; ++d) bkofs[d] = lhalf * 16 + 2 * d;

    v8f acc[8];
#pragma unroll
    for (int t = 0; t < 8; ++t) {
        float bv = bias[t * 16 + l16];      // bias depends on N only
#pragma unroll
        for (int r = 0; r < 8; ++r) acc[t][r] = bv;
    }

    const unsigned short* rowA = A + (size_t)(mbase + l16) * As;

    for (int k0 = 0; k0 < Kp; k0 += 32) {
        v8u au;
#pragma unroll
        for (int d = 0; d < 8; ++d)
            au[d] = *(const unsigned int*)(rowA + k0 + akofs[d]);
        v16bf av = __builtin_bit_cast(v16bf, au);

#pragma unroll
        for (int t = 0; t < 8; ++t) {
            const unsigned short* rowB = Bw + (size_t)(t * 16 + l16) * Kp;
            v8u bu;
#pragma unroll
            for (int d = 0; d < 8; ++d)
                bu[d] = *(const unsigned int*)(rowB + k0 + bkofs[d]);
            v16bf bv = __builtin_bit_cast(v16bf, bu);
            acc[t] = __builtin_amdgcn_wmma_f32_16x16x32_bf16(
                false, av, false, bv, (short)0, acc[t], false, false);
        }
    }

    const int mrow = mbase + lhalf * 8;     // C layout: VGPR r -> M = mbase + lhalf*8 + r
    if (MODE == 0) {
        float* O = (float*)out;
#pragma unroll
        for (int t = 0; t < 8; ++t) {
            int n = t * 16 + l16;
#pragma unroll
            for (int r = 0; r < 8; ++r)
                O[(size_t)(mrow + r) * HIDN + n] = acc[t][r];
        }
    } else {
        unsigned short* O = (unsigned short*)out;
#pragma unroll
        for (int t = 0; t < 8; ++t) {
            int n = t * 16 + l16;
#pragma unroll
            for (int r = 0; r < 8; ++r) {
                float v = acc[t][r];
                v = v > 0.f ? v : 0.f;
                O[(size_t)(mrow + r) * ostride + ocol + n] = f2bf(v);
            }
        }
    }
}

// ---------------------------------------------------------------------------
// One-time NCHW -> NHWC transpose of x (coalesced writes, L2-resident after).
// ---------------------------------------------------------------------------
__global__ void nhwc_k(const float* __restrict__ x, float* __restrict__ ximg)
{
    int i = blockIdx.x * blockDim.x + threadIdx.x;      // B*H*W*C = 3,211,264
    if (i >= BB * HH * WW * CC) return;
    int c = i & (CC - 1);
    int t = i >> 8;                                     // CC == 256
    int xq = t % WW; t /= WW;
    int y  = t % HH;
    int b  = t / HH;
    ximg[i] = x[(((size_t)b * CC + c) * HH + y) * WW + xq];
}

// ---------------------------------------------------------------------------
// Bilinear vert-align from NHWC image: block per vertex, thread = channel.
// Lanes read consecutive channels -> fully coalesced 4-tap gathers.
// ---------------------------------------------------------------------------
__global__ void vert_align_k(const float* __restrict__ ximg,
                             const float* __restrict__ verts,
                             unsigned short* __restrict__ featA)
{
    int vt = blockIdx.x;
    int c  = threadIdx.x;
    int b  = vt / VV;
    float vx = verts[vt * 3 + 0], vy = verts[vt * 3 + 1];
    float px = fminf(fmaxf((vx + 1.f) * (WW - 1) * 0.5f, 0.f), (float)(WW - 1));
    float py = fminf(fmaxf((vy + 1.f) * (HH - 1) * 0.5f, 0.f), (float)(HH - 1));
    int x0 = (int)floorf(px), y0 = (int)floorf(py);
    int x1 = min(x0 + 1, WW - 1), y1 = min(y0 + 1, HH - 1);
    float wx = px - (float)x0, wy = py - (float)y0;
    const float* ib = ximg + (size_t)b * HH * WW * CC;
    float t00 = ib[((size_t)y0 * WW + x0) * CC + c];
    float t01 = ib[((size_t)y0 * WW + x1) * CC + c];
    float t10 = ib[((size_t)y1 * WW + x0) * CC + c];
    float t11 = ib[((size_t)y1 * WW + x1) * CC + c];
    float top = (1.f - wx) * t00 + wx * t01;
    float bot = (1.f - wx) * t10 + wx * t11;
    featA[(size_t)vt * CC + c] = f2bf((1.f - wy) * top + wy * bot);
}

// fp32 weight (N x K) -> bf16 (N x Kp), zero padded
__global__ void convw_k(const float* __restrict__ src, unsigned short* __restrict__ dst,
                        int K, int Kp, int total)
{
    int i = blockIdx.x * blockDim.x + threadIdx.x;
    if (i >= total) return;
    int n = i / Kp, k = i - n * Kp;
    dst[i] = (k < K) ? f2bf(src[(size_t)n * K + k]) : (unsigned short)0;
}

// ---------------------------------------------------------------------------
// Edge scatter, coalesced: thread = (directed edge, channel). 128 consecutive
// lanes hit 128 consecutive floats for both the read and the atomic add.
// ---------------------------------------------------------------------------
__global__ void scatter_k(const int* __restrict__ edges,
                          const float* __restrict__ vw1,
                          float* __restrict__ pre)
{
    long i = (long)blockIdx.x * blockDim.x + threadIdx.x;   // 2E*128 threads
    if (i >= (long)2 * EE * HIDN) return;
    int  c = (int)(i & (HIDN - 1));
    long e = i >> 7;                                        // HIDN == 128
    int a, bI;
    if (e < EE) { a = edges[2 * e];                 bI = edges[2 * e + 1]; }
    else        { long ee = e - EE; a = edges[2 * ee + 1]; bI = edges[2 * ee]; }
    atomicAdd(pre + (size_t)a * HIDN + c, vw1[(size_t)bI * HIDN + c]);
}

// relu(pre) -> vf[:,0:128] bf16; verts -> vf[:,128:131]; zeros -> vf[:,131:160]
__global__ void relu_pack_k(const float* __restrict__ pre,
                            const float* __restrict__ verts,
                            unsigned short* __restrict__ vf)
{
    long i = (long)blockIdx.x * blockDim.x + threadIdx.x;
    if (i >= (long)VT_ * 160) return;
    int vt = (int)(i / 160), j = (int)(i - (long)vt * 160);
    unsigned short v;
    if (j < 128)      { float f = pre[(size_t)vt * HIDN + j]; v = f2bf(f > 0.f ? f : 0.f); }
    else if (j < 131) { v = f2bf(verts[vt * 3 + (j - 128)]); }
    else              { v = 0; }
    vf[(size_t)vt * 288 + j] = v;
}

// write verts cols + zero pad tail after the img GEMM
__global__ void pack_tail_k(unsigned short* __restrict__ vf,
                            const float* __restrict__ verts, int colv, int zend)
{
    int vt = blockIdx.x * blockDim.x + threadIdx.x;
    if (vt >= VT_) return;
    unsigned short* r = vf + (size_t)vt * 288;
    r[colv + 0] = f2bf(verts[vt * 3 + 0]);
    r[colv + 1] = f2bf(verts[vt * 3 + 1]);
    r[colv + 2] = f2bf(verts[vt * 3 + 2]);
    for (int j = colv + 3; j < zend; ++j) r[j] = 0;
}

// deform = tanh(vf[:, :131] @ Wo^T + bo); verts += deform; emit stage output
__global__ void deform_k(const unsigned short* __restrict__ vf,
                         const float* __restrict__ Wo, const float* __restrict__ bo,
                         float* __restrict__ verts, float* __restrict__ outp)
{
    int vt = blockIdx.x * blockDim.x + threadIdx.x;
    if (vt >= VT_) return;
    const unsigned short* r = vf + (size_t)vt * 288;
#pragma unroll
    for (int j = 0; j < 3; ++j) {
        const float* w = Wo + j * 131;
        float s = bo[j];
        for (int k = 0; k < 131; ++k) s += bf2f(r[k]) * w[k];
        float nv = verts[vt * 3 + j] + tanhf(s);
        verts[vt * 3 + j] = nv;
        outp[vt * 3 + j]  = nv;
    }
}

extern "C" void kernel_launch(void* const* d_in, const int* in_sizes, int n_in,
                              void* d_out, int out_size, void* d_ws, size_t ws_size,
                              hipStream_t stream)
{
    (void)in_sizes; (void)n_in; (void)out_size; (void)ws_size;
    const float* x     = (const float*)d_in[0];
    const int*   edges = (const int*)d_in[2];

    char* ws = (char*)d_ws;
    size_t off = 0;
    auto alloc = [&](size_t bytes) -> void* {
        void* p = ws + off;
        off = (off + bytes + 255) & ~(size_t)255;
        return p;
    };
    unsigned short* vf    = (unsigned short*)alloc((size_t)VT_ * 288 * 2); // bf16 features, K-padded
    unsigned short* featA = (unsigned short*)alloc((size_t)VT_ * 256 * 2); // sampled feats / aliased vw1
    float*          vw1   = (float*)featA;                                  // fp32 VT x 128 (same size)
    float*          pre   = (float*)alloc((size_t)VT_ * HIDN * 4);
    float*          verts = (float*)alloc((size_t)VT_ * 3 * 4);
    float*          ximg  = (float*)alloc((size_t)BB * HH * WW * CC * 4);   // NHWC image
    unsigned short* wbslt = (unsigned short*)alloc((size_t)128 * 256 * 2);
    unsigned short* w0slt = (unsigned short*)alloc((size_t)128 * 288 * 2);
    unsigned short* w1slt = (unsigned short*)alloc((size_t)128 * 288 * 2);

    hipMemcpyAsync(verts, d_in[1], (size_t)VT_ * 3 * 4, hipMemcpyDeviceToDevice, stream);
    nhwc_k<<<(BB * HH * WW * CC + 255) / 256, 256, 0, stream>>>(x, ximg);

    float* outp = (float*)d_out;
    const int MBLK = VT_ / 128;            // 1281, exact

    for (int s = 0; s < 3; ++s) {
        int pi = 3 + s * 16;               // Wb, bb, 3x(w0,b0,w1,b1), Wo, bo
        const float* Wb = (const float*)d_in[pi + 0];
        const float* bb = (const float*)d_in[pi + 1];
        const float* Wo = (const float*)d_in[pi + 14];
        const float* bo = (const float*)d_in[pi + 15];

        // sample image features at current verts (coalesced NHWC gathers)
        vert_align_k<<<VT_, 256, 0, stream>>>(ximg, verts, featA);
        // img = relu(feat @ Wb^T + bb) -> bf16 into vf (cols 0..127 stage0, 128..255 later)
        convw_k<<<(128 * 256 + 255) / 256, 256, 0, stream>>>(Wb, wbslt, 256, 256, 128 * 256);
        gemm_bf16_wmma<1><<<MBLK, 256, 0, stream>>>(featA, 256, wbslt, 256, bb,
                                                    (void*)vf, 288, s == 0 ? 0 : 128);
        pack_tail_k<<<(VT_ + 255) / 256, 256, 0, stream>>>(vf, verts,
                                                           s == 0 ? 128 : 256,
                                                           s == 0 ? 160 : 288);

        for (int g = 0; g < 3; ++g) {
            int base = pi + 2 + g * 4;
            const float* w0 = (const float*)d_in[base + 0];
            const float* b0 = (const float*)d_in[base + 1];
            const float* w1 = (const float*)d_in[base + 2];
            const float* b1 = (const float*)d_in[base + 3];
            int K  = (s > 0 && g == 0) ? 259 : 131;
            int Kp = (K == 259) ? 288 : 160;

            convw_k<<<(128 * Kp + 255) / 256, 256, 0, stream>>>(w0, w0slt, K, Kp, 128 * Kp);
            convw_k<<<(128 * Kp + 255) / 256, 256, 0, stream>>>(w1, w1slt, K, Kp, 128 * Kp);
            // vw1 = vf @ w1^T + b1 ; pre = vf @ w0^T + b0
            gemm_bf16_wmma<0><<<MBLK, 256, 0, stream>>>(vf, 288, w1slt, Kp, b1, (void*)vw1, 0, 0);
            gemm_bf16_wmma<0><<<MBLK, 256, 0, stream>>>(vf, 288, w0slt, Kp, b0, (void*)pre, 0, 0);
            // pre += segment_sum(vw1 over edges, both directions), coalesced atomics
            scatter_k<<<(int)(((long)2 * EE * HIDN + 255) / 256), 256, 0, stream>>>(edges, vw1, pre);
            // vf = [relu(pre) | verts | 0-pad] (K=131, Kp=160 for next GEMM)
            relu_pack_k<<<(int)(((size_t)VT_ * 160 + 255) / 256), 256, 0, stream>>>(pre, verts, vf);
        }
        // verts += tanh(vf @ Wo^T + bo); emit stage output
        deform_k<<<(VT_ + 255) / 256, 256, 0, stream>>>(vf, Wo, bo, verts,
                                                        outp + (size_t)s * VT_ * 3);
    }
}
